// MGAFE_8907762172242
// MI455X (gfx1250) — matrix operand order repaired and checked by reading, not compiled, not run on hardware
//
#include <hip/hip_runtime.h>

// ---------------------------------------------------------------------------
// MI455X (gfx1250) implementation. wave32, WMMA f32_16x16x32_f16 for every
// GEMM; flash-style attention with a *fixed* softmax shift (legal because
// q,k are L2-normalized => |score| <= |temperature|), row sums computed by a
// ones-matrix WMMA, and K/V tiles double-buffered into LDS with
// GLOBAL_LOAD_ASYNC_TO_LDS_B128 (ASYNCcnt) to overlap HBM/L2 latency with
// matrix math. Fragment addressing follows CDNA5 ISA 7.12.2.
// ---------------------------------------------------------------------------

typedef _Float16 half_t;
typedef __attribute__((ext_vector_type(16))) _Float16 v16h;
typedef __attribute__((ext_vector_type(8)))  _Float16 v8h;
typedef __attribute__((ext_vector_type(8)))  float    v8f;

#define DEV __device__ __forceinline__

DEV v16h ld16(const half_t* p0, const half_t* p1) {
    v16h r;
    *reinterpret_cast<v8h*>(&r)       = *reinterpret_cast<const v8h*>(p0);
    *(reinterpret_cast<v8h*>(&r) + 1) = *reinterpret_cast<const v8h*>(p1);
    return r;
}

DEV v8f wmma_f16(v16h a, v16h b, v8f c) {
    return __builtin_amdgcn_wmma_f32_16x16x32_f16(
        /*neg_a=*/false, a, /*neg_b=*/false, b,
        /*c_mod=*/(short)0, c, /*reuse_a=*/false, /*reuse_b=*/false);
}

// Async DMA: 16B per lane, global -> LDS, tracked by ASYNCcnt.
DEV void async_cp16(unsigned int lds_off, const void* gp) {
    unsigned long long ga = (unsigned long long)gp;
    asm volatile("global_load_async_to_lds_b128 %0, %1, off"
                 :: "v"(lds_off), "v"(ga) : "memory");
}

// ---------------------------- elementwise prep -----------------------------

__global__ void cvt_f32_f16(const float* __restrict__ in, half_t* __restrict__ out, int n) {
    int i = blockIdx.x * blockDim.x + threadIdx.x;
    if (i < n) out[i] = (half_t)in[i];
}

// OIHW [co][ci][kh][kw] fp32  ->  [co][(kh*KS+kw)*256 + ci] f16
template<int KS>
__global__ void reorder_w(const float* __restrict__ src, half_t* __restrict__ dst) {
    int t = blockIdx.x * blockDim.x + threadIdx.x;     // 256*KS*KS*256 threads
    const int row = KS * KS * 256;
    int co = t / row;
    int r  = t - co * row;
    int s  = r >> 8;           // spatial tap
    int ci = r & 255;
    int kh = s / KS, kw = s % KS;
    dst[t] = (half_t)src[((co * 256 + ci) * KS + kh) * KS + kw];
}

// --------------------- implicit-GEMM conv via WMMA -------------------------
// src: f16 [4096][256] (pixel-major).  wt: f16 [256][KS*KS*256].
// CMAJOR: out fp32 [256][4096] (at outp); else out fp32 [4096][256].
template<int KS, bool CMAJOR>
__global__ __launch_bounds__(128) void conv_wmma(const half_t* __restrict__ src,
                                                 const half_t* __restrict__ wt,
                                                 float* __restrict__ outp) {
    const int lane   = threadIdx.x & 31;
    const int wid    = blockIdx.x * (blockDim.x >> 5) + (threadIdx.x >> 5);
    const int ptile  = wid >> 4;          // 256 pixel tiles
    const int cotile = wid & 15;          // 16 out-channel tiles
    const int pbase  = ptile * 16;
    const int cobase = cotile * 16;
    const int l15 = lane & 15;
    const int lhi = lane >> 4;            // 0: lanes 0-15, 1: lanes 16-31
    const int kb  = lhi * 8;

    const int pix = pbase + l15;          // A-matrix row (M) for this lane
    const int py = pix >> 6, px = pix & 63;
    const int wrowlen = KS * KS * 256;
    const half_t* wrow = wt + (size_t)(cobase + l15) * wrowlen + lhi * 8;

    v8f acc = {};
    constexpr int PAD = KS / 2;
    for (int s = 0; s < KS * KS; ++s) {
        const int dy = s / KS - PAD, dx = s % KS - PAD;
        const int sy = py + dy, sx = px + dx;
        const bool valid = ((unsigned)sy < 64u) && ((unsigned)sx < 64u);
        const half_t* arow = src + (size_t)(valid ? (sy * 64 + sx) : 0) * 256;
        __builtin_prefetch(wrow + (s + 1) * 256, 0, 0);
        #pragma unroll
        for (int cc = 0; cc < 8; ++cc) {
            const int k0 = cc * 32;
            v16h a = {};
            if (valid) {
                *reinterpret_cast<v8h*>(&a) =
                    *reinterpret_cast<const v8h*>(arow + k0 + kb);
                *(reinterpret_cast<v8h*>(&a) + 1) =
                    *reinterpret_cast<const v8h*>(arow + k0 + 16 + kb);
            }
            v16h b = ld16(wrow + s * 256 + k0, wrow + s * 256 + k0 + 16);
            acc = wmma_f16(a, b, acc);
        }
    }
    // C/D layout: lanes 0-15: (M=r, N=lane); lanes 16-31: (M=8+r, N=lane-16)
    if (CMAJOR) {
        float* op = outp + (size_t)(cobase + l15) * 4096 + pbase + lhi * 8;
        #pragma unroll
        for (int r = 0; r < 8; ++r) op[r] = acc[r];
    } else {
        float* op = outp + (size_t)(pbase + lhi * 8) * 256 + cobase + l15;
        #pragma unroll
        for (int r = 0; r < 8; ++r) op[(size_t)r * 256] = acc[r];
    }
}

// ------------------- depthwise 3x3 + head split/transpose ------------------
// qkv_pre fp32 [768][4096] -> q,k: f16 [8][4096][32]; v: f16 [8][32][4096]
__global__ void dwconv_split(const float* __restrict__ qkv,
                             const float* __restrict__ wdw,
                             half_t* __restrict__ qh, half_t* __restrict__ kh,
                             half_t* __restrict__ vt) {
    int tid = blockIdx.x * blockDim.x + threadIdx.x;   // 768*4096
    int c3 = tid >> 12;
    int p  = tid & 4095;
    int y = p >> 6, x = p & 63;
    const float* w  = wdw + c3 * 9;
    const float* iv = qkv + (size_t)c3 * 4096;
    float s = 0.f;
    #pragma unroll
    for (int dy = -1; dy <= 1; ++dy)
        #pragma unroll
        for (int dx = -1; dx <= 1; ++dx) {
            int yy = y + dy, xx = x + dx;
            if ((unsigned)yy < 64u && (unsigned)xx < 64u)
                s += iv[yy * 64 + xx] * w[(dy + 1) * 3 + (dx + 1)];
        }
    int grp = c3 >> 8;            // 0=q 1=k 2=v
    int c = c3 & 255;
    int head = c >> 5, ch = c & 31;
    half_t hv = (half_t)s;
    if (grp == 0)      qh[((size_t)head * 4096 + p) * 32 + ch] = hv;
    else if (grp == 1) kh[((size_t)head * 4096 + p) * 32 + ch] = hv;
    else               vt[((size_t)head * 32 + ch) * 4096 + p] = hv;
}

// L2-normalize each length-32 row (torch F.normalize w/ eps=1e-12)
__global__ void l2norm32(half_t* __restrict__ t) {   // [8*4096][32]
    int v = blockIdx.x * blockDim.x + threadIdx.x;
    half_t* r = t + (size_t)v * 32;
    float vals[32], ss = 0.f;
    #pragma unroll
    for (int i = 0; i < 32; ++i) { vals[i] = (float)r[i]; ss += vals[i] * vals[i]; }
    float inv = 1.f / fmaxf(sqrtf(ss), 1e-12f);
    #pragma unroll
    for (int i = 0; i < 32; ++i) r[i] = (half_t)(vals[i] * inv);
}

// ----------------------------- flash attention -----------------------------
// One wave per (head, 16-query tile). d=32 -> one K=32 WMMA per 16x16 S tile.
// Fixed softmax shift (|s| <= |temp| since q,k unit vectors); row sums via
// ones-matrix WMMA; K/V double-buffered in LDS via async b128 DMA.
//
// Per-wave LDS map (halfs): [0,2048)  K bufs (2 x 32key x 32ch)
//                           [2048,4096) V bufs (2 x 32ch x 32key)
//                           [4096,4608) P tile (16 x 32)
__global__ __launch_bounds__(128) void attn_flash(const half_t* __restrict__ qh,
                                                  const half_t* __restrict__ kh,
                                                  const half_t* __restrict__ vt,
                                                  const float* __restrict__ temp,
                                                  half_t* __restrict__ oh) {
    __shared__ half_t smem[4][4608];
    const int lane  = threadIdx.x & 31;
    const int wslot = threadIdx.x >> 5;
    const int wid   = blockIdx.x * 4 + wslot;     // 2048 waves total
    const int head  = wid >> 8;
    const int qt    = wid & 255;
    const int l15 = lane & 15, lhi = lane >> 4, kb = lhi * 8;

    half_t* wbase = &smem[wslot][0];
    const unsigned int ldsBase = (unsigned int)(unsigned long long)wbase; // LDS byte offset

    const half_t* qrow = qh + ((size_t)head * 4096 + qt * 16 + l15) * 32;
    const v16h aQ = ld16(qrow + lhi * 16, qrow + lhi * 16 + 8);
    const float tscale = temp[head];
    const float shift  = fabsf(tscale);   // |q.k| <= 1  =>  exp(s-shift) <= ~1

    v16h bOne;
    #pragma unroll
    for (int i = 0; i < 16; ++i) bOne[i] = (half_t)1.0f;

    v8f o0 = {}, o1 = {}, lacc = {};

    const char* kgp = (const char*)(kh + (size_t)head * 4096 * 32);  // [key][32ch]
    const char* vgp = (const char*)(vt + (size_t)head * 32 * 4096);  // [ch][4096key]

    // issue async copy of chunk kt into buffer b (8 x b128 = 4KB)
    auto issue = [&](int kt, int b) {
        // K chunk: 2048 contiguous bytes
        const char* gk = kgp + (size_t)kt * 2048 + lane * 16;
        unsigned int lk = ldsBase + b * 2048 + lane * 16;
        #pragma unroll
        for (int j = 0; j < 4; ++j) async_cp16(lk + j * 512, gk + j * 512);
        // V chunk: 32 rows (ch) x 64B, global row stride 8192B
        const char* gv = vgp + (size_t)kt * 64 + (size_t)lane * 8192;
        unsigned int lv = ldsBase + 4096 + b * 2048 + lane * 64;
        #pragma unroll
        for (int j = 0; j < 4; ++j) async_cp16(lv + j * 16, gv + j * 16);
    };

    issue(0, 0);
    half_t* pP = wbase + 4096;

    for (int kt = 0; kt < 128; ++kt) {            // 32 keys per chunk
        const int b = kt & 1;
        if (kt + 1 < 128) {
            issue(kt + 1, b ^ 1);
            asm volatile("s_wait_asynccnt 0x8" ::: "memory"); // current buf done
        } else {
            asm volatile("s_wait_asynccnt 0x0" ::: "memory");
        }
        const half_t* kbuf = wbase + b * 1024;          // [key][32]
        const half_t* vbuf = wbase + 2048 + b * 1024;   // [ch][32]

        v16h bK0 = ld16(kbuf + l15 * 32 + lhi * 16,        kbuf + l15 * 32 + lhi * 16 + 8);
        v16h bK1 = ld16(kbuf + (16 + l15) * 32 + lhi * 16, kbuf + (16 + l15) * 32 + lhi * 16 + 8);
        v8f z = {};
        v8f S0 = wmma_f16(aQ, bK0, z);
        v8f S1 = wmma_f16(aQ, bK1, z);

        #pragma unroll
        for (int r = 0; r < 8; ++r) {
            float p0 = __expf(S0[r] * tscale - shift);
            float p1 = __expf(S1[r] * tscale - shift);
            int row = lhi * 8 + r;                 // C-layout row for this reg
            pP[row * 32 + l15]      = (half_t)p0;  // keys 0-15 of chunk
            pP[row * 32 + 16 + l15] = (half_t)p1;  // keys 16-31
        }
        asm volatile("s_wait_dscnt 0x0" ::: "memory");   // cross-lane LDS RAW

        const half_t* prow = pP + l15 * 32;
        v16h aP = ld16(prow + lhi * 16, prow + lhi * 16 + 8);
        v16h bV0 = ld16(vbuf + l15 * 32 + lhi * 16,        vbuf + l15 * 32 + lhi * 16 + 8);
        v16h bV1 = ld16(vbuf + (16 + l15) * 32 + lhi * 16, vbuf + (16 + l15) * 32 + lhi * 16 + 8);
        o0   = wmma_f16(aP, bV0, o0);
        o1   = wmma_f16(aP, bV1, o1);
        lacc = wmma_f16(aP, bOne, lacc);   // row sums, replicated per column
    }
    #pragma unroll
    for (int r = 0; r < 8; ++r) {
        float invl = 1.f / lacc[r];
        int row = qt * 16 + lhi * 8 + r;
        half_t* op = oh + (size_t)row * 256 + head * 32;
        op[l15]      = (half_t)(o0[r] * invl);
        op[16 + l15] = (half_t)(o1[r] * invl);
    }
}

// ------------------------------- launcher ----------------------------------

extern "C" void kernel_launch(void* const* d_in, const int* in_sizes, int n_in,
                              void* d_out, int out_size, void* d_ws, size_t ws_size,
                              hipStream_t stream) {
    const float* x     = (const float*)d_in[0];   // (1,64,64,256) NHWC = [4096][256]
    const float* W1    = (const float*)d_in[1];
    const float* W3    = (const float*)d_in[2];
    const float* W5    = (const float*)d_in[3];
    const float* Wdw   = (const float*)d_in[4];
    const float* Wproj = (const float*)d_in[5];
    const float* temp  = (const float*)d_in[6];
    float* out = (float*)d_out;

    // workspace carve-out
    char* ws = (char*)d_ws;
    size_t off = 0;
    auto carve = [&](size_t bytes) { void* p = ws + off; off = (off + bytes + 255) & ~(size_t)255; return p; };
    half_t* x_h  = (half_t*)carve((size_t)4096 * 256 * 2);
    half_t* w1t  = (half_t*)carve((size_t)256 * 256 * 2);
    half_t* w3t  = (half_t*)carve((size_t)256 * 2304 * 2);
    half_t* w5t  = (half_t*)carve((size_t)256 * 6400 * 2);
    half_t* wpt  = (half_t*)carve((size_t)256 * 256 * 2);
    float*  qkv  = (float*)carve((size_t)768 * 4096 * 4);
    half_t* qh   = (half_t*)carve((size_t)8 * 4096 * 32 * 2);
    half_t* khh  = (half_t*)carve((size_t)8 * 4096 * 32 * 2);
    half_t* vtt  = (half_t*)carve((size_t)8 * 32 * 4096 * 2);
    half_t* oh   = (half_t*)carve((size_t)4096 * 256 * 2);
    (void)ws_size; (void)n_in; (void)in_sizes; (void)out_size;

    // 1. f16 conversion / weight reorder
    cvt_f32_f16<<<4096, 256, 0, stream>>>(x, x_h, 4096 * 256);
    reorder_w<1><<<256, 256, 0, stream>>>(W1, w1t);
    reorder_w<3><<<2304, 256, 0, stream>>>(W3, w3t);
    reorder_w<5><<<6400, 256, 0, stream>>>(W5, w5t);
    reorder_w<1><<<256, 256, 0, stream>>>(Wproj, wpt);

    // 2. multi-scale QKV convs (implicit GEMM, WMMA) -> qkv [768][4096] fp32
    conv_wmma<1, true><<<1024, 128, 0, stream>>>(x_h, w1t, qkv);
    conv_wmma<3, true><<<1024, 128, 0, stream>>>(x_h, w3t, qkv + (size_t)256 * 4096);
    conv_wmma<5, true><<<1024, 128, 0, stream>>>(x_h, w5t, qkv + (size_t)512 * 4096);

    // 3. depthwise 3x3 + split into head layouts (v transposed)
    dwconv_split<<<12288, 256, 0, stream>>>(qkv, Wdw, qh, khh, vtt);

    // 4. L2 normalize q, k rows
    l2norm32<<<128, 256, 0, stream>>>(qh);
    l2norm32<<<128, 256, 0, stream>>>(khh);

    // 5. flash attention (8 heads x 256 query tiles = 2048 waves)
    attn_flash<<<512, 128, 0, stream>>>(qh, khh, vtt, temp, oh);

    // 6. 1x1 projection GEMM -> fp32 NHWC output
    conv_wmma<1, false><<<1024, 128, 0, stream>>>(oh, wpt, out);
}
